// MultiHeadSelfAttention_65721589563789
// MI455X (gfx1250) — compile-verified, matrix operand order checked by reading, not run on hardware
//
#include <hip/hip_runtime.h>

// ---------------------------------------------------------------------------
// MHSA forward for gfx1250 (MI455X): f16 WMMA everywhere, flash attention,
// async global->LDS staging of K tiles (ASYNCcnt path).
// B=2, T=2048, D=1024, H=16, DK=64.
// ---------------------------------------------------------------------------

typedef __attribute__((ext_vector_type(16))) _Float16 v16h;
typedef __attribute__((ext_vector_type(8)))  _Float16 v8h;
typedef __attribute__((ext_vector_type(4)))  _Float16 v4h;
typedef __attribute__((ext_vector_type(8)))  float    v8f;
typedef __attribute__((ext_vector_type(4)))  int      int4v;

typedef __attribute__((address_space(1))) int4v* gptr_b128;
typedef __attribute__((address_space(3))) int4v* lptr_b128;

#define WMMA_F16(A, B, C) \
    __builtin_amdgcn_wmma_f32_16x16x32_f16(false, (A), false, (B), (short)0, (C), false, false)

#define ASYNC_LD_B128(gsrc, ldst) \
    __builtin_amdgcn_global_load_async_to_lds_b128( \
        (gptr_b128)(int4v*)(gsrc), (lptr_b128)(int4v*)(ldst), 0, 0)

// Load a 16x32 f16 fragment (A or B operand) from a row-major matrix.
// Lane L: row = row0 + (L%16); halves 0..7  = [k0 + (L/16)*8 .. +7]
//                              halves 8..15 = [k0 + (L/16)*8 + 16 .. +7]
__device__ __forceinline__ v16h load_frag(const _Float16* __restrict__ base,
                                          int row0, int ld, int k0, int lane) {
    int r  = row0 + (lane & 15);
    int kb = k0 + ((lane >> 4) << 3);
    const _Float16* p = base + (size_t)r * ld + kb;
    v8h lo = *(const v8h*)(p);
    v8h hi = *(const v8h*)(p + 16);
    return __builtin_shufflevector(lo, hi, 0,1,2,3,4,5,6,7,8,9,10,11,12,13,14,15);
}

// ---------------------------------------------------------------------------
// fp32 -> f16 conversion (vector of 4)
// ---------------------------------------------------------------------------
__global__ void cvt_f32_to_f16(const float* __restrict__ in,
                               _Float16* __restrict__ out, int n4) {
    int i = blockIdx.x * blockDim.x + threadIdx.x;
    if (i < n4) {
        const float4 v = ((const float4*)in)[i];
        v4h o;
        o[0] = (_Float16)v.x; o[1] = (_Float16)v.y;
        o[2] = (_Float16)v.z; o[3] = (_Float16)v.w;
        ((v4h*)out)[i] = o;
    }
}

// ---------------------------------------------------------------------------
// QKV projection: C[4096,3072] = xh[4096,1024] @ Wqkv[3072,1024]^T
// Each wave: 32m x 64n tile. Block = 8 waves (2m x 4n) -> 64m x 256n.
// Epilogue scatters f16 results into per-head Q/K/V buffers [B*H][T][DK].
// ---------------------------------------------------------------------------
__global__ __launch_bounds__(256) void qkv_gemm(const _Float16* __restrict__ xh,
                                                const _Float16* __restrict__ wh,
                                                _Float16* __restrict__ Qh,
                                                _Float16* __restrict__ Kh,
                                                _Float16* __restrict__ Vh) {
    const int lane = threadIdx.x & 31;
    const int wave = threadIdx.x >> 5;
    const int m0 = blockIdx.x * 64 + (wave & 1) * 32;
    const int n0 = blockIdx.y * 256 + (wave >> 1) * 64;

    v8f c[2][4] = {};
    for (int k0 = 0; k0 < 1024; k0 += 32) {
        v16h a0 = load_frag(xh, m0,      1024, k0, lane);
        v16h a1 = load_frag(xh, m0 + 16, 1024, k0, lane);
        v16h b0 = load_frag(wh, n0,      1024, k0, lane);
        v16h b1 = load_frag(wh, n0 + 16, 1024, k0, lane);
        v16h b2 = load_frag(wh, n0 + 32, 1024, k0, lane);
        v16h b3 = load_frag(wh, n0 + 48, 1024, k0, lane);
        c[0][0] = WMMA_F16(a0, b0, c[0][0]);
        c[0][1] = WMMA_F16(a0, b1, c[0][1]);
        c[0][2] = WMMA_F16(a0, b2, c[0][2]);
        c[0][3] = WMMA_F16(a0, b3, c[0][3]);
        c[1][0] = WMMA_F16(a1, b0, c[1][0]);
        c[1][1] = WMMA_F16(a1, b1, c[1][1]);
        c[1][2] = WMMA_F16(a1, b2, c[1][2]);
        c[1][3] = WMMA_F16(a1, b3, c[1][3]);
    }

    _Float16* const dsts[3] = {Qh, Kh, Vh};
#pragma unroll
    for (int i = 0; i < 2; ++i) {
#pragma unroll
        for (int j = 0; j < 4; ++j) {
            const int nb    = n0 + j * 16;          // 16-aligned, within 64-block
            const int which = nb >> 10;             // 0:Q 1:K 2:V (uniform per frag)
            const int h     = (nb >> 6) & 15;       // head (uniform per frag)
            const int dk    = (nb & 63) + (lane & 15);
            _Float16* dst   = dsts[which];
#pragma unroll
            for (int r = 0; r < 8; ++r) {
                const int m = m0 + i * 16 + r + ((lane >> 4) << 3);
                const int b = m >> 11;              // batch
                const int t = m & 2047;             // time
                dst[(size_t)(((b << 4) + h) * 2048 + t) * 64 + dk] = (_Float16)c[i][j][r];
            }
        }
    }
}

// ---------------------------------------------------------------------------
// Flash attention (causal). Block = 4 waves = 64 q rows for one (b,h).
// K tile is async-staged into LDS once per block (GLOBAL_LOAD_ASYNC_TO_LDS,
// ASYNCcnt) and shared by all 4 waves; V is transposed through LDS.
// Wave computes S^T = K @ Q^T (per-lane softmax), then O^T = V^T @ P^T with
// P^T fed straight from S^T C-fragments into the B operand (in registers).
// ---------------------------------------------------------------------------
__global__ __launch_bounds__(128) void flash_attn(const _Float16* __restrict__ Qh,
                                                  const _Float16* __restrict__ Kh,
                                                  const _Float16* __restrict__ Vh,
                                                  _Float16* __restrict__ Oc) {
    __shared__ __align__(16) _Float16 klds[32 * 72]; // K tile [key][dk], padded rows
    __shared__ __align__(16) _Float16 vt[64 * 40];   // V^T tile [dk][key], padded

    const int tid  = threadIdx.x;
    const int lane = tid & 31;
    const int wave = tid >> 5;
    const int bh   = blockIdx.y;                     // b*16 + h
    const int q0   = blockIdx.x * 64;
    const int qw0  = q0 + wave * 16;

    const _Float16* Qb = Qh + (size_t)bh * 2048 * 64;
    const _Float16* Kb = Kh + (size_t)bh * 2048 * 64;
    const _Float16* Vb = Vh + (size_t)bh * 2048 * 64;

    // Persistent Q B-operand fragments (16 q rows x DK=64 -> two k-steps)
    const v16h qf0 = load_frag(Qb, qw0, 64, 0,  lane);
    const v16h qf1 = load_frag(Qb, qw0, 64, 32, lane);

    v8f   ot[4] = {};                 // O^T accum: d = dd*16 + r + 8*(lane/16), q = lane%16
    float m_run = -1.0e30f;
    float l_run = 0.0f;
    const float scale = 0.125f;       // 1/sqrt(64)
    const int   kend  = q0 + 64;      // causal bound for this block

    for (int kt = 0; kt < kend; kt += 32) {
        // --- async-stage K[kt..kt+32) row-major into LDS (2 x B128 per thread) ---
        {
            const int ch0 = tid;                    // chunks of 8 halves (16B)
            const int ch1 = tid + 128;              // 256 chunks total (4KB)
            const int i0 = ch0 >> 3, s0 = (ch0 & 7) << 3;
            const int i1 = ch1 >> 3, s1 = (ch1 & 7) << 3;
            ASYNC_LD_B128(Kb + (size_t)(kt + i0) * 64 + s0, &klds[i0 * 72 + s0]);
            ASYNC_LD_B128(Kb + (size_t)(kt + i1) * 64 + s1, &klds[i1 * 72 + s1]);
        }
        // --- cooperative: stage V[kt..kt+32) transposed into LDS ---
        {
            const int i  = tid >> 2;            // key row 0..31
            const int d0 = (tid & 3) << 4;      // dk sub-block 0/16/32/48
            const _Float16* src = Vb + (size_t)(kt + i) * 64 + d0;
            v8h x0 = *(const v8h*)(src);
            v8h x1 = *(const v8h*)(src + 8);
#pragma unroll
            for (int j = 0; j < 8; ++j) {
                vt[(d0 + j)     * 40 + i] = x0[j];
                vt[(d0 + 8 + j) * 40 + i] = x1[j];
            }
        }
        __builtin_amdgcn_s_wait_asynccnt(0);
        __syncthreads();

        if (kt < qw0 + 16) {          // wave-uniform: this wave needs this key tile
            __builtin_prefetch(Vb + (size_t)(kt + 32) * 64, 0, 1);

            // S^T tiles: keys [kt,kt+16) and [kt+16,kt+32), 16 q cols (K from LDS)
            v8f z = {};
            v16h ka = load_frag(klds, 0, 72, 0, lane);
            v8f s0  = WMMA_F16(ka, qf0, z);
            ka      = load_frag(klds, 0, 72, 32, lane);
            s0      = WMMA_F16(ka, qf1, s0);
            ka      = load_frag(klds, 16, 72, 0, lane);
            v8f s1  = WMMA_F16(ka, qf0, z);
            ka      = load_frag(klds, 16, 72, 32, lane);
            s1      = WMMA_F16(ka, qf1, s1);

            // scale + causal mask (only needed on/near the diagonal)
            const int q       = qw0 + (lane & 15);
            const int keybase = kt + ((lane >> 4) << 3);
            if (kt + 32 > qw0) {
#pragma unroll
                for (int r = 0; r < 8; ++r) {
                    const int k0i = keybase + r;
                    s0[r] = (k0i      <= q) ? s0[r] * scale : -__builtin_inff();
                    s1[r] = (k0i + 16 <= q) ? s1[r] * scale : -__builtin_inff();
                }
            } else {
#pragma unroll
                for (int r = 0; r < 8; ++r) { s0[r] *= scale; s1[r] *= scale; }
            }

            // online softmax: per-lane (fixed q) + one xor-16 combine
            float mnew = m_run;
#pragma unroll
            for (int r = 0; r < 8; ++r)
                mnew = fmaxf(mnew, fmaxf(s0[r], s1[r]));
            mnew = fmaxf(mnew, __shfl_xor(mnew, 16, 32));

            const float alpha = __expf(m_run - mnew);
            float lsum = 0.0f;
#pragma unroll
            for (int r = 0; r < 8; ++r) {
                const float p0 = __expf(s0[r] - mnew);
                const float p1 = __expf(s1[r] - mnew);
                s0[r] = p0; s1[r] = p1;
                lsum += p0 + p1;
            }
            lsum  = lsum + __shfl_xor(lsum, 16, 32);
            l_run = l_run * alpha + lsum;
            m_run = mnew;

            // P^T -> B operand directly in registers (layouts coincide)
            v16h pb;
#pragma unroll
            for (int r = 0; r < 8; ++r) {
                pb[r]     = (_Float16)s0[r];
                pb[r + 8] = (_Float16)s1[r];
            }

            // O^T = O^T*alpha + V^T @ P^T
#pragma unroll
            for (int dd = 0; dd < 4; ++dd) {
#pragma unroll
                for (int r = 0; r < 8; ++r) ot[dd][r] *= alpha;
                v16h av = load_frag(vt, dd * 16, 40, 0, lane);  // V^T from LDS
                ot[dd]  = WMMA_F16(av, pb, ot[dd]);
            }
        }
        __syncthreads();
    }

    // epilogue: normalize and store f16 into Oc[b][t][h*64 + d]
    const float linv = 1.0f / l_run;
    const int   q    = qw0 + (lane & 15);
    const int   b    = bh >> 4;
    const int   h    = bh & 15;
    _Float16* orow = Oc + ((size_t)(b * 2048 + q)) * 1024 + h * 64;
#pragma unroll
    for (int dd = 0; dd < 4; ++dd) {
#pragma unroll
        for (int r = 0; r < 8; ++r) {
            const int d = dd * 16 + r + ((lane >> 4) << 3);
            orow[d] = (_Float16)(ot[dd][r] * linv);
        }
    }
}

// ---------------------------------------------------------------------------
// Output projection: out[4096,1024] = Oc[4096,1024] @ Wo[1024,1024]^T  (fp32 out)
// ---------------------------------------------------------------------------
__global__ __launch_bounds__(256) void out_gemm(const _Float16* __restrict__ oh,
                                                const _Float16* __restrict__ wo,
                                                float* __restrict__ out) {
    const int lane = threadIdx.x & 31;
    const int wave = threadIdx.x >> 5;
    const int m0 = blockIdx.x * 64 + (wave & 1) * 32;
    const int n0 = blockIdx.y * 256 + (wave >> 1) * 64;

    v8f c[2][4] = {};
    for (int k0 = 0; k0 < 1024; k0 += 32) {
        v16h a0 = load_frag(oh, m0,      1024, k0, lane);
        v16h a1 = load_frag(oh, m0 + 16, 1024, k0, lane);
        v16h b0 = load_frag(wo, n0,      1024, k0, lane);
        v16h b1 = load_frag(wo, n0 + 16, 1024, k0, lane);
        v16h b2 = load_frag(wo, n0 + 32, 1024, k0, lane);
        v16h b3 = load_frag(wo, n0 + 48, 1024, k0, lane);
        c[0][0] = WMMA_F16(a0, b0, c[0][0]);
        c[0][1] = WMMA_F16(a0, b1, c[0][1]);
        c[0][2] = WMMA_F16(a0, b2, c[0][2]);
        c[0][3] = WMMA_F16(a0, b3, c[0][3]);
        c[1][0] = WMMA_F16(a1, b0, c[1][0]);
        c[1][1] = WMMA_F16(a1, b1, c[1][1]);
        c[1][2] = WMMA_F16(a1, b2, c[1][2]);
        c[1][3] = WMMA_F16(a1, b3, c[1][3]);
    }
#pragma unroll
    for (int i = 0; i < 2; ++i) {
#pragma unroll
        for (int j = 0; j < 4; ++j) {
            const int n = n0 + j * 16 + (lane & 15);
#pragma unroll
            for (int r = 0; r < 8; ++r) {
                const int m = m0 + i * 16 + r + ((lane >> 4) << 3);
                out[(size_t)m * 1024 + n] = c[i][j][r];
            }
        }
    }
}

// ---------------------------------------------------------------------------
extern "C" void kernel_launch(void* const* d_in, const int* in_sizes, int n_in,
                              void* d_out, int out_size, void* d_ws, size_t ws_size,
                              hipStream_t stream) {
    const float* x    = (const float*)d_in[0];   // [2,2048,1024]
    const float* wqkv = (const float*)d_in[1];   // [3072,1024]
    const float* wo   = (const float*)d_in[2];   // [1024,1024]
    float*       out  = (float*)d_out;           // [2,2048,1024]

    _Float16* xh  = (_Float16*)d_ws;                 // 4096*1024
    _Float16* wqh = xh  + (size_t)4096 * 1024;       // 3072*1024
    _Float16* woh = wqh + (size_t)3072 * 1024;       // 1024*1024
    _Float16* Qh  = woh + (size_t)1024 * 1024;       // 32*2048*64
    _Float16* Kh  = Qh  + (size_t)32 * 2048 * 64;
    _Float16* Vh  = Kh  + (size_t)32 * 2048 * 64;
    _Float16* Oc  = Vh  + (size_t)32 * 2048 * 64;    // 4096*1024

    const int nx = 4096 * 1024 / 4, nq = 3072 * 1024 / 4, no = 1024 * 1024 / 4;
    cvt_f32_to_f16<<<(nx + 255) / 256, 256, 0, stream>>>(x,    xh,  nx);
    cvt_f32_to_f16<<<(nq + 255) / 256, 256, 0, stream>>>(wqkv, wqh, nq);
    cvt_f32_to_f16<<<(no + 255) / 256, 256, 0, stream>>>(wo,   woh, no);

    qkv_gemm<<<dim3(64, 12), 256, 0, stream>>>(xh, wqh, Qh, Kh, Vh);
    flash_attn<<<dim3(32, 32), 128, 0, stream>>>(Qh, Kh, Vh, Oc);
    out_gemm<<<dim3(64, 4), 256, 0, stream>>>(Oc, woh, out);
}
